// MambaForecaster_52295521796340
// MI455X (gfx1250) — matrix operand order repaired and checked by reading, not compiled, not run on hardware
//
#include <hip/hip_runtime.h>
#include <hip/hip_bf16.h>
#include <math.h>

// ---------------------------------------------------------------------------
// MambaForecaster for MI455X (gfx1250): bf16 WMMA GEMMs (async-LDS double
// buffered) + wave-per-channel selective scan
// ---------------------------------------------------------------------------
#define BB 4
#define TT 1024
#define NROW (BB * TT)        // 4096
#define DMODEL 512
#define DINNER 1024
#define DSTATE 64
#define DTRANK 32
#define DPROJ  160            // DT_RANK + 2*D_STATE
#define DPROJP 256            // padded to a multiple of 128 -> no GEMM edge path
#define NLAYERS 2
#define HORIZON 24
#define DOUT 8

typedef __bf16 v16bf __attribute__((ext_vector_type(16)));
typedef __bf16 bf8v  __attribute__((ext_vector_type(8)));
typedef float  v8f   __attribute__((ext_vector_type(8)));

// Async global->LDS 16-byte copy (VGLOBAL async encoding, ASYNCcnt-tracked).
// VDST carries the LDS byte address (generic LDS pointers keep the offset in
// the low 32 bits; aperture lives in the high bits).
__device__ __forceinline__ void async_b128(const void* g, void* l) {
  unsigned lds = (unsigned)(unsigned long long)l;
  unsigned long long ga = (unsigned long long)g;
  asm volatile("global_load_async_to_lds_b128 %0, %1, off"
               :: "v"(lds), "v"(ga) : "memory");
}
__device__ __forceinline__ void wait_async0() {
  asm volatile("s_wait_asynccnt 0x0" ::: "memory");
}

// --------------------------- f32 -> bf16 convert ---------------------------
__global__ void k_cvt_bf16(const float* __restrict__ s, __bf16* __restrict__ d, int n) {
  int i = blockIdx.x * blockDim.x + threadIdx.x;
  if (i < n) d[i] = (__bf16)s[i];
}
// convert with column padding: src [rows][nc] -> dst [rows][ncp], pad = 0
__global__ void k_cvt_pad(const float* __restrict__ s, __bf16* __restrict__ d,
                          int rows, int nc, int ncp) {
  int i = blockIdx.x * blockDim.x + threadIdx.x;   // rows*ncp
  if (i >= rows * ncp) return;
  int r = i / ncp, c = i - r * ncp;
  d[i] = (c < nc) ? (__bf16)s[r * nc + c] : (__bf16)0.0f;
}

// --------------------------- input projection ------------------------------
__global__ void k_inproj(const float* __restrict__ x, const float* __restrict__ Win,
                         const float* __restrict__ bin,
                         float* __restrict__ h, __bf16* __restrict__ hb) {
  int idx = blockIdx.x * blockDim.x + threadIdx.x;   // NROW*DMODEL
  int row = idx >> 9, j = idx & 511;
  float acc = bin[j];
  #pragma unroll
  for (int k = 0; k < 8; ++k) acc += x[row * 8 + k] * Win[k * DMODEL + j];
  h[idx] = acc;
  hb[idx] = (__bf16)acc;
}

// --------------------------- WMMA bf16 GEMM --------------------------------
// C[M,N] = A[M,K] @ B[K,N] (+bias) (+softplus). Row-major bf16 A/B, f32 C,
// optional bf16 mirror. Block 128x128, 8 waves of 64x32, K-step 32.
// M, N multiples of 128; K multiple of 32 (guaranteed by caller).
__global__ __launch_bounds__(256)
void k_gemm_bf16(const __bf16* __restrict__ A, const __bf16* __restrict__ B,
                 float* __restrict__ C, __bf16* __restrict__ Cb,
                 const float* __restrict__ bias, int M, int N, int K, int act) {
  __shared__ __bf16 As[2][128][40];    // 80B rows: 16B aligned, 2-phase banks
  __shared__ __bf16 Bs[2][32][136];    // 272B rows: 16B aligned, 2-phase banks

  const int tid  = threadIdx.x;
  const int lane = tid & 31;
  const int wave = tid >> 5;
  const int wm = wave >> 2;            // 0..1 (64-row slab)
  const int wn = wave & 3;             // 0..3 (32-col slab)
  const int half = lane >> 4;
  const int r16  = lane & 15;
  const int bm = blockIdx.y * 128;
  const int bn = blockIdx.x * 128;

  auto load_tiles = [&](int kt, int buf) {
    { // A tile 128x32: each thread 16 contiguous bf16 = 2 async b128
      int rr = tid >> 1, cc = (tid & 1) * 16;
      const __bf16* g = &A[(size_t)(bm + rr) * K + kt * 32 + cc];
      async_b128(g,     &As[buf][rr][cc]);
      async_b128(g + 8, &As[buf][rr][cc + 8]);
    }
    { // B tile 32x128
      int rr = tid >> 3, cc = (tid & 7) * 16;
      const __bf16* g = &B[(size_t)(kt * 32 + rr) * N + bn + cc];
      async_b128(g,     &Bs[buf][rr][cc]);
      async_b128(g + 8, &Bs[buf][rr][cc + 8]);
    }
  };

  v8f zero = {};
  v8f acc[4][2];
  #pragma unroll
  for (int mi = 0; mi < 4; ++mi)
    #pragma unroll
    for (int ni = 0; ni < 2; ++ni) acc[mi][ni] = zero;

  const int kTiles = K >> 5;
  load_tiles(0, 0);
  wait_async0();
  __syncthreads();

  for (int kt = 0; kt < kTiles; ++kt) {
    const int cur = kt & 1;
    if (kt + 1 < kTiles) load_tiles(kt + 1, cur ^ 1);  // overlap with WMMA

    // A fragment: lane half selects K-subgroup; two contiguous 8-elem chunks
    v16bf afr[4];
    #pragma unroll
    for (int mi = 0; mi < 4; ++mi) {
      int ar = wm * 64 + mi * 16 + r16;
      const __bf16* ap = &As[cur][ar][8 * half];
      bf8v lo = *(const bf8v*)ap;          // K 0..7  (+8 if upper half)
      bf8v hi = *(const bf8v*)(ap + 16);   // K 16..23 (+8 if upper half)
      afr[mi] = __builtin_shufflevector(lo, hi, 0,1,2,3,4,5,6,7,8,9,10,11,12,13,14,15);
    }
    // B fragment: lane l holds column N=l%16, K in same order as A fragment
    v16bf bfr[2];
    #pragma unroll
    for (int ni = 0; ni < 2; ++ni) {
      const int bc = wn * 32 + ni * 16 + r16;
      #pragma unroll
      for (int s = 0; s < 16; ++s)
        bfr[ni][s] = Bs[cur][(s < 8 ? 8 * half + s : 8 + 8 * half + s)][bc];
    }

    #pragma unroll
    for (int mi = 0; mi < 4; ++mi)
      #pragma unroll
      for (int ni = 0; ni < 2; ++ni)
        acc[mi][ni] = __builtin_amdgcn_wmma_f32_16x16x32_bf16(
            false, afr[mi], false, bfr[ni], (short)0, acc[mi][ni], false, false);

    if (kt + 1 < kTiles) wait_async0();
    __syncthreads();
  }

  // Epilogue: C/D layout -> VGPR i holds row i (lanes 0-15) / i+8 (lanes 16-31)
  #pragma unroll
  for (int mi = 0; mi < 4; ++mi)
    #pragma unroll
    for (int ni = 0; ni < 2; ++ni) {
      int col = bn + wn * 32 + ni * 16 + r16;
      float bv = bias ? bias[col] : 0.0f;
      #pragma unroll
      for (int i = 0; i < 8; ++i) {
        int rowg = bm + wm * 64 + mi * 16 + i + 8 * half;
        float v = acc[mi][ni][i] + bv;
        if (act == 1)  // branch-free softplus: max(x,0) + log(1+exp(-|x|))
          v = fmaxf(v, 0.0f) + __logf(1.0f + __expf(-fabsf(v)));
        C[(size_t)rowg * N + col] = v;
        if (Cb) Cb[(size_t)rowg * N + col] = (__bf16)v;
      }
    }
}

// --------------------- causal depthwise conv4 + SiLU -----------------------
__global__ void k_conv(const float* __restrict__ xz, const float* __restrict__ cw,
                       const float* __restrict__ cb,
                       float* __restrict__ xc, __bf16* __restrict__ xcb) {
  int idx = blockIdx.x * blockDim.x + threadIdx.x;   // NROW*DINNER
  int c = idx & (DINNER - 1);
  int row = idx >> 10;
  int t = row & (TT - 1);
  float acc = cb[c];
  #pragma unroll
  for (int j = 0; j < 4; ++j) {
    int tt = t - 3 + j;
    if (tt >= 0) acc += cw[c * 4 + j] * xz[(size_t)(row - 3 + j) * (2 * DINNER) + c];
  }
  float s = acc / (1.0f + __expf(-acc));   // SiLU
  xc[idx] = s;
  xcb[idx] = (__bf16)s;
}

// --------------------- dt_r slice -> bf16 ----------------------------------
__global__ void k_dtr(const float* __restrict__ xdb, __bf16* __restrict__ dtrb) {
  int idx = blockIdx.x * blockDim.x + threadIdx.x;   // NROW*DTRANK
  dtrb[idx] = (__bf16)xdb[(size_t)(idx >> 5) * DPROJP + (idx & 31)];
}

// --------------------- selective scan: one wave per (b,d) ------------------
__global__ __launch_bounds__(256)
void k_scan(const float* __restrict__ Alog, const float* __restrict__ dt,
            const float* __restrict__ xc, const float* __restrict__ xdb,
            float* __restrict__ y) {
  int lane = threadIdx.x & 31;
  int w = blockIdx.x * 8 + (threadIdx.x >> 5);   // 0..4095
  int b = w >> 10;
  int d = w & (DINNER - 1);

  float A0 = -__expf(Alog[d * DSTATE + lane]);
  float A1 = -__expf(Alog[d * DSTATE + lane + 32]);
  float h0 = 0.0f, h1 = 0.0f;

  for (int t = 0; t < TT; ++t) {
    int row = b * TT + t;
    float dtv = dt[(size_t)row * DINNER + d];
    float xv  = xc[(size_t)row * DINNER + d];
    const float* p = &xdb[(size_t)row * DPROJP];
    float Bv0 = p[DTRANK + lane],      Bv1 = p[DTRANK + 32 + lane];
    float Cv0 = p[DTRANK + 64 + lane], Cv1 = p[DTRANK + 96 + lane];
    float dA0 = __expf(dtv * A0), dA1 = __expf(dtv * A1);
    float u = dtv * xv;
    h0 = dA0 * h0 + u * Bv0;
    h1 = dA1 * h1 + u * Bv1;
    float yv = h0 * Cv0 + h1 * Cv1;
    #pragma unroll
    for (int off = 16; off > 0; off >>= 1) yv += __shfl_xor(yv, off, 32);
    if (lane == 0) y[(size_t)row * DINNER + d] = yv;
  }
}

// --------------------- skip + gate combine ---------------------------------
__global__ void k_combine(const float* __restrict__ xz, const float* __restrict__ y,
                          const float* __restrict__ xc, const float* __restrict__ Dsk,
                          __bf16* __restrict__ ycb) {
  int idx = blockIdx.x * blockDim.x + threadIdx.x;   // NROW*DINNER
  int c = idx & (DINNER - 1);
  int row = idx >> 10;
  float z = xz[(size_t)row * (2 * DINNER) + DINNER + c];
  float v = (y[idx] + Dsk[c] * xc[idx]) * (z / (1.0f + __expf(-z)));
  ycb[idx] = (__bf16)v;
}

// --------------------- LayerNorm on last timestep only ---------------------
__global__ void k_ln_last(const float* __restrict__ h, const float* __restrict__ g,
                          const float* __restrict__ be, float* __restrict__ out) {
  __shared__ float red[256];
  int b = blockIdx.x, tid = threadIdx.x;
  const float* rowp = h + (size_t)(b * TT + (TT - 1)) * DMODEL;
  float s = 0.0f;
  for (int j = tid; j < DMODEL; j += 256) s += rowp[j];
  red[tid] = s; __syncthreads();
  for (int o = 128; o > 0; o >>= 1) { if (tid < o) red[tid] += red[tid + o]; __syncthreads(); }
  float mu = red[0] / (float)DMODEL; __syncthreads();
  float vv = 0.0f;
  for (int j = tid; j < DMODEL; j += 256) { float dd = rowp[j] - mu; vv += dd * dd; }
  red[tid] = vv; __syncthreads();
  for (int o = 128; o > 0; o >>= 1) { if (tid < o) red[tid] += red[tid + o]; __syncthreads(); }
  float inv = rsqrtf(red[0] / (float)DMODEL + 1e-5f);
  for (int j = tid; j < DMODEL; j += 256)
    out[b * DMODEL + j] = (rowp[j] - mu) * inv * g[j] + be[j];
}

// --------------------- head (tiny f32 GEMMs) -------------------------------
__global__ void k_head1(const float* __restrict__ ln, const float* __restrict__ W,
                        const float* __restrict__ bia, float* __restrict__ hid) {
  int b = blockIdx.x, j = threadIdx.x;   // 512 threads
  float acc = bia[j];
  for (int k = 0; k < DMODEL; ++k) acc += ln[b * DMODEL + k] * W[k * DMODEL + j];
  hid[b * DMODEL + j] = 0.5f * acc * (1.0f + erff(acc * 0.70710678118f));  // exact GELU
}
__global__ void k_head2(const float* __restrict__ hid, const float* __restrict__ W,
                        const float* __restrict__ bia, float* __restrict__ out) {
  int b = blockIdx.x, j = threadIdx.x;   // 192 threads
  float acc = bia[j];
  for (int k = 0; k < DMODEL; ++k) acc += hid[b * DMODEL + k] * W[k * (HORIZON * DOUT) + j];
  out[b * (HORIZON * DOUT) + j] = acc;
}

// ---------------------------------------------------------------------------
extern "C" void kernel_launch(void* const* d_in, const int* in_sizes, int n_in,
                              void* d_out, int out_size, void* d_ws, size_t ws_size,
                              hipStream_t stream) {
  (void)in_sizes; (void)n_in; (void)out_size; (void)ws_size;
  const float* x      = (const float*)d_in[0];
  const float* W_in   = (const float*)d_in[1];
  const float* b_in   = (const float*)d_in[2];
  const float* W_xz   = (const float*)d_in[3];
  const float* conv_w = (const float*)d_in[4];
  const float* conv_b = (const float*)d_in[5];
  const float* W_xp   = (const float*)d_in[6];
  const float* W_dt   = (const float*)d_in[7];
  const float* b_dt   = (const float*)d_in[8];
  const float* A_log  = (const float*)d_in[9];
  const float* D_skip = (const float*)d_in[10];
  const float* W_out  = (const float*)d_in[11];
  const float* ln_g   = (const float*)d_in[12];
  const float* ln_b   = (const float*)d_in[13];
  const float* W_h1   = (const float*)d_in[14];
  const float* b_h1   = (const float*)d_in[15];
  const float* W_h2   = (const float*)d_in[16];
  const float* b_h2   = (const float*)d_in[17];
  float* out = (float*)d_out;

  char* ws = (char*)d_ws;
  size_t off = 0;
  auto take = [&](size_t bytes) { void* p = ws + off; off += (bytes + 255) & ~(size_t)255; return p; };

  float*  h_f   = (float*) take((size_t)NROW * DMODEL * 4);
  __bf16* h_b   = (__bf16*)take((size_t)NROW * DMODEL * 2);
  float*  xz_f  = (float*) take((size_t)NROW * 2 * DINNER * 4);
  float*  xc_f  = (float*) take((size_t)NROW * DINNER * 4);
  __bf16* xc_b  = (__bf16*)take((size_t)NROW * DINNER * 2);
  float*  xdb_f = (float*) take((size_t)NROW * DPROJP * 4);
  __bf16* dtr_b = (__bf16*)take((size_t)NROW * DTRANK * 2);
  float*  dt_f  = (float*) take((size_t)NROW * DINNER * 4);
  float*  y_f   = (float*) take((size_t)NROW * DINNER * 4);
  __bf16* yc_b  = (__bf16*)take((size_t)NROW * DINNER * 2);
  __bf16* wxz_b = (__bf16*)take((size_t)NLAYERS * DMODEL * 2 * DINNER * 2);
  __bf16* wxp_b = (__bf16*)take((size_t)NLAYERS * DINNER * DPROJP * 2);
  __bf16* wdt_b = (__bf16*)take((size_t)NLAYERS * DTRANK * DINNER * 2);
  __bf16* wout_b= (__bf16*)take((size_t)NLAYERS * DINNER * DMODEL * 2);
  float*  ln_o  = (float*) take((size_t)BB * DMODEL * 4);
  float*  hid_f = (float*) take((size_t)BB * DMODEL * 4);

  // weights -> bf16 (W_xproj zero-padded 160 -> 256 cols)
  int nwxz = NLAYERS * DMODEL * 2 * DINNER;
  int nwxpP = NLAYERS * DINNER * DPROJP;
  int nwdt = NLAYERS * DTRANK * DINNER;
  int nwo  = NLAYERS * DINNER * DMODEL;
  k_cvt_bf16<<<(nwxz + 255) / 256, 256, 0, stream>>>(W_xz,  wxz_b,  nwxz);
  k_cvt_pad<<<(nwxpP + 255) / 256, 256, 0, stream>>>(W_xp, wxp_b,
      NLAYERS * DINNER, DPROJ, DPROJP);
  k_cvt_bf16<<<(nwdt + 255) / 256, 256, 0, stream>>>(W_dt,  wdt_b,  nwdt);
  k_cvt_bf16<<<(nwo  + 255) / 256, 256, 0, stream>>>(W_out, wout_b, nwo);

  // input projection
  k_inproj<<<(NROW * DMODEL) / 256, 256, 0, stream>>>(x, W_in, b_in, h_f, h_b);

  for (int l = 0; l < NLAYERS; ++l) {
    // xz = h @ W_xz
    k_gemm_bf16<<<dim3((2 * DINNER) / 128, NROW / 128), 256, 0, stream>>>(
        h_b, wxz_b + (size_t)l * DMODEL * 2 * DINNER, xz_f, nullptr, nullptr,
        NROW, 2 * DINNER, DMODEL, 0);
    // causal conv + SiLU
    k_conv<<<(NROW * DINNER) / 256, 256, 0, stream>>>(
        xz_f, conv_w + (size_t)l * DINNER * 4, conv_b + (size_t)l * DINNER, xc_f, xc_b);
    // xdb = xc @ W_xproj (padded N=256, no edge path)
    k_gemm_bf16<<<dim3(DPROJP / 128, NROW / 128), 256, 0, stream>>>(
        xc_b, wxp_b + (size_t)l * DINNER * DPROJP, xdb_f, nullptr, nullptr,
        NROW, DPROJP, DINNER, 0);
    // dt_r slice -> bf16
    k_dtr<<<(NROW * DTRANK) / 256, 256, 0, stream>>>(xdb_f, dtr_b);
    // dt = softplus(dt_r @ W_dt + b_dt) (fused epilogue)
    k_gemm_bf16<<<dim3(DINNER / 128, NROW / 128), 256, 0, stream>>>(
        dtr_b, wdt_b + (size_t)l * DTRANK * DINNER, dt_f, nullptr,
        b_dt + (size_t)l * DINNER, NROW, DINNER, DTRANK, 1);
    // selective scan
    k_scan<<<(BB * DINNER) / 8, 256, 0, stream>>>(
        A_log + (size_t)l * DINNER * DSTATE, dt_f, xc_f, xdb_f, y_f);
    // y = (y + D*xc) * silu(z)
    k_combine<<<(NROW * DINNER) / 256, 256, 0, stream>>>(
        xz_f, y_f, xc_f, D_skip + (size_t)l * DINNER, yc_b);
    // h = y @ W_out (f32 + bf16 mirror for next layer)
    k_gemm_bf16<<<dim3(DMODEL / 128, NROW / 128), 256, 0, stream>>>(
        yc_b, wout_b + (size_t)l * DINNER * DMODEL, h_f, h_b, nullptr,
        NROW, DMODEL, DINNER, 0);
  }

  // LN (last timestep rows only) + head
  k_ln_last<<<BB, 256, 0, stream>>>(h_f, ln_g, ln_b, ln_o);
  k_head1<<<BB, DMODEL, 0, stream>>>(ln_o, W_h1, b_h1, hid_f);
  k_head2<<<BB, HORIZON * DOUT, 0, stream>>>(hid_f, W_h2, b_h2, out);
}
